// MultiHeadedAttention_27986006901073
// MI455X (gfx1250) — matrix operand order repaired and checked by reading, not compiled, not run on hardware
//
#include <hip/hip_runtime.h>
#include <hip/hip_bf16.h>

// Problem constants (from reference): D_MODEL=1024, H=16, Dk=64, B=2, S=2048.
#define DM   1024
#define NH   16
#define DKH  64
#define SEQ  2048

typedef __attribute__((ext_vector_type(16))) __bf16 bf16x16;
typedef __attribute__((ext_vector_type(8)))  float  f32x8;
typedef __attribute__((ext_vector_type(4)))  int    v4i;
typedef __attribute__((address_space(1))) v4i v4i_g;   // global
typedef __attribute__((address_space(3))) v4i v4i_s;   // LDS

struct U4 { unsigned x, y, z, w; };
struct U8 { U4 a, b; };

// Load two 16-byte chunks (8 bf16 each) and fuse into one 32-byte WMMA fragment.
static __device__ __forceinline__ bf16x16 ldpair(const __bf16* p0, const __bf16* p1) {
  U8 u;
  u.a = *reinterpret_cast<const U4*>(p0);
  u.b = *reinterpret_cast<const U4*>(p1);
  return __builtin_bit_cast(bf16x16, u);
}

static __device__ __forceinline__ f32x8 wmma_bf16(bf16x16 a, bf16x16 b, f32x8 c) {
  // (neg_a, A, neg_b, B, c_mod, C, reuse_a, reuse_b)
  return __builtin_amdgcn_wmma_f32_16x16x32_bf16(false, a, false, b, (short)0, c,
                                                 false, false);
}

// XOR-lane swap within each 16-lane row via DPP ROW_XMASK (VALU, no LDS traffic).
template <int MASK>
static __device__ __forceinline__ float dpp_xor(float v) {
  int i = __builtin_bit_cast(int, v);
  int r = __builtin_amdgcn_update_dpp(i, i, 0x160 | MASK, 0xf, 0xf, true);
  return __builtin_bit_cast(float, r);
}
static __device__ __forceinline__ float rowmax16(float v) {
  v = fmaxf(v, dpp_xor<1>(v));
  v = fmaxf(v, dpp_xor<2>(v));
  v = fmaxf(v, dpp_xor<4>(v));
  v = fmaxf(v, dpp_xor<8>(v));
  return v;
}
static __device__ __forceinline__ float rowsum16(float v) {
  v += dpp_xor<1>(v);
  v += dpp_xor<2>(v);
  v += dpp_xor<4>(v);
  v += dpp_xor<8>(v);
  return v;
}

// ---- CDNA5 async global->LDS staging (ASYNCcnt-tracked, no VGPR data) -----
#if __has_builtin(__builtin_amdgcn_global_load_async_to_lds_b128)
#define HAS_ASYNC_LDS 1
static __device__ __forceinline__ void async_cp16(const __bf16* g, __bf16* l) {
  __builtin_amdgcn_global_load_async_to_lds_b128((v4i_g*)g, (v4i_s*)l, 0, 0);
}
#else
#define HAS_ASYNC_LDS 0
#endif

static __device__ __forceinline__ void wait_async0() {
#if HAS_ASYNC_LDS
#if __has_builtin(__builtin_amdgcn_s_wait_asynccnt)
  __builtin_amdgcn_s_wait_asynccnt(0);
#else
  asm volatile("s_wait_asynccnt 0" ::: "memory");
#endif
#endif
}

// ---------------------------------------------------------------------------
// One-shot fp32 -> bf16 conversion (bandwidth-bound; removes per-tile cvt).
// ---------------------------------------------------------------------------
struct B4 { __bf16 a, b, c, d; };
__global__ __launch_bounds__(256) void cvt_kernel(const float* __restrict__ in,
                                                  __bf16* __restrict__ out, int n4) {
  const int i = blockIdx.x * blockDim.x + threadIdx.x;
  if (i < n4) {
    const float4 v = reinterpret_cast<const float4*>(in)[i];
    B4 o;
    o.a = (__bf16)v.x; o.b = (__bf16)v.y; o.c = (__bf16)v.z; o.d = (__bf16)v.w;
    reinterpret_cast<B4*>(out)[i] = o;
  }
}

// ---------------------------------------------------------------------------
// GEMM: C[M,DM] = X[M,DM] @ W[DM,DM]^T + bias.  bf16 inputs, fp32 accum.
// 32(M) x 64(N) tile per wave: 2 A-frags x 4 B-frags -> 8 WMMA per k-step.
// MODE 0: store bf16 as [B,H,S,Dk]  (Q, K)
// MODE 1: store bf16 as [B,H,Dk,S]  (V transposed)
// MODE 2: store fp32 as [M,DM]      (output projection)
// ---------------------------------------------------------------------------
template <int MODE>
__global__ __launch_bounds__(256) void gemm_kernel(
    const __bf16* __restrict__ X, const __bf16* __restrict__ W,
    const float* __restrict__ bias, __bf16* __restrict__ obf,
    float* __restrict__ of32) {
  const int lane = threadIdx.x & 31;
  const int wid  = (blockIdx.x * blockDim.x + threadIdx.x) >> 5;
  const int ntiles = DM / 64;              // 16
  const int tm = (wid / ntiles) * 32;
  const int tn = (wid % ntiles) * 64;
  const int l16 = lane & 15, h = lane >> 4;

  const __bf16* xr0 = X + (size_t)(tm + l16) * DM;
  const __bf16* xr1 = X + (size_t)(tm + 16 + l16) * DM;
  const __bf16* wr0 = W + (size_t)(tn + l16) * DM;
  const __bf16* wr1 = W + (size_t)(tn + 16 + l16) * DM;
  const __bf16* wr2 = W + (size_t)(tn + 32 + l16) * DM;
  const __bf16* wr3 = W + (size_t)(tn + 48 + l16) * DM;

  f32x8 acc[2][4] = {};
#pragma unroll 2
  for (int k = 0; k < DM; k += 32) {
    const bf16x16 a0 = ldpair(xr0 + k + h * 16, xr0 + k + h * 16 + 8);
    const bf16x16 a1 = ldpair(xr1 + k + h * 16, xr1 + k + h * 16 + 8);
    const bf16x16 b0 = ldpair(wr0 + k + h * 16, wr0 + k + h * 16 + 8);
    const bf16x16 b1 = ldpair(wr1 + k + h * 16, wr1 + k + h * 16 + 8);
    const bf16x16 b2 = ldpair(wr2 + k + h * 16, wr2 + k + h * 16 + 8);
    const bf16x16 b3 = ldpair(wr3 + k + h * 16, wr3 + k + h * 16 + 8);
    acc[0][0] = wmma_bf16(a0, b0, acc[0][0]);
    acc[1][0] = wmma_bf16(a1, b0, acc[1][0]);
    acc[0][1] = wmma_bf16(a0, b1, acc[0][1]);
    acc[1][1] = wmma_bf16(a1, b1, acc[1][1]);
    acc[0][2] = wmma_bf16(a0, b2, acc[0][2]);
    acc[1][2] = wmma_bf16(a1, b2, acc[1][2]);
    acc[0][3] = wmma_bf16(a0, b3, acc[0][3]);
    acc[1][3] = wmma_bf16(a1, b3, acc[1][3]);
  }

#pragma unroll
  for (int j = 0; j < 4; ++j) {
    const int ncol = tn + j * 16 + l16;     // C layout: lane holds one column
    const float bn = bias[ncol];
#pragma unroll
    for (int i = 0; i < 2; ++i) {
#pragma unroll
      for (int r = 0; r < 8; ++r) {         // row = r + 8*(lane/16)
        const float v = acc[i][j][r] + bn;
        const int row = tm + i * 16 + r + 8 * h;
        if constexpr (MODE == 2) {
          of32[(size_t)row * DM + ncol] = v;
        } else {
          const int bb = row / SEQ, s = row % SEQ;
          const int hd = ncol >> 6, d = ncol & 63;
          if constexpr (MODE == 0)
            obf[(((size_t)(bb * NH + hd)) * SEQ + s) * DKH + d] = (__bf16)v;
          else
            obf[(((size_t)(bb * NH + hd)) * DKH + d) * SEQ + s] = (__bf16)v;
        }
      }
    }
  }
}

// ---------------------------------------------------------------------------
// Flash attention. Block = 8 waves = 128 Q rows of one head. Each 32-key block
// of K (32x64) and V^T (64x32) is staged once into a double-buffered LDS tile
// (async global->LDS when available) and consumed by all 8 waves.
// Q,K: [B,H,S,Dk] bf16.  Vt: [B,H,Dk,S] bf16.  X2 out: [B,S,H*Dk] bf16.
// ---------------------------------------------------------------------------
__global__ __launch_bounds__(256) void attn_kernel(
    const __bf16* __restrict__ Q, const __bf16* __restrict__ K,
    const __bf16* __restrict__ Vt, __bf16* __restrict__ X2) {
  __shared__ __bf16 kt[2][32 * 72];         // 32 keys x 64 d, pad->72
  __shared__ __bf16 vt[2][64 * 40];         // 64 d x 32 keys, pad->40
  __shared__ __bf16 pstage[8][16 * 40];     // per-wave P tile

  const int tid  = threadIdx.x;
  const int lane = tid & 31;
  const int wl   = tid >> 5;
  const int bh   = blockIdx.x / (SEQ / 128);   // batch*H + head
  const int qblk = blockIdx.x % (SEQ / 128);
  const int qt   = qblk * 8 + wl;              // wave's 16-row q tile
  const int l16 = lane & 15, h = lane >> 4;

  const __bf16* Qh = Q  + (size_t)bh * SEQ * DKH;
  const __bf16* Kh = K  + (size_t)bh * SEQ * DKH;
  const __bf16* Vh = Vt + (size_t)bh * DKH * SEQ;

  // Q A-fragments (d=0..31, d=32..63), loaded once per wave.
  const __bf16* qp = Qh + (size_t)(qt * 16 + l16) * DKH;
  const bf16x16 aq0 = ldpair(qp + h * 16,      qp + h * 16 + 8);
  const bf16x16 aq1 = ldpair(qp + 32 + h * 16, qp + 32 + h * 16 + 8);

  // Cooperative staging coordinates (one 16B chunk per thread per tile).
  const int krow = tid >> 3, kc = tid & 7;     // K: 32 rows x 8 chunks
  const int vrow = tid >> 2, vc = tid & 3;     // V: 64 rows x 4 chunks

  auto stage = [&](int kb, int buf) {
    __bf16*       kdst = &kt[buf][krow * 72 + kc * 8];
    const __bf16* ksrc = Kh + (size_t)(kb + krow) * DKH + kc * 8;
    __bf16*       vdst = &vt[buf][vrow * 40 + vc * 8];
    const __bf16* vsrc = Vh + (size_t)vrow * SEQ + kb + vc * 8;
#if HAS_ASYNC_LDS
    async_cp16(ksrc, kdst);
    async_cp16(vsrc, vdst);
#else
    *reinterpret_cast<U4*>(kdst) = *reinterpret_cast<const U4*>(ksrc);
    *reinterpret_cast<U4*>(vdst) = *reinterpret_cast<const U4*>(vsrc);
#endif
  };

  f32x8 oacc[4] = {};
  float mst[8], lst[8];
#pragma unroll
  for (int r = 0; r < 8; ++r) { mst[r] = -1e30f; lst[r] = 0.0f; }
  __bf16* pst = &pstage[wl][0];
  const float scale = 0.125f;                  // 1/sqrt(Dk)

  stage(0, 0);                                 // prologue fill of buffer 0
  wait_async0();
  __syncthreads();

  for (int kb = 0; kb < SEQ; kb += 32) {
    const int buf = (kb >> 5) & 1;
    if (kb + 32 < SEQ) stage(kb + 32, buf ^ 1);  // overlap next-block staging

    // Scores S = Q @ K^T : two 16(q) x 16(key) C tiles, K-dim = Dk = 64.
    const __bf16* k0 = &kt[buf][l16 * 72];
    const __bf16* k1 = &kt[buf][(16 + l16) * 72];
    const bf16x16 b00 = ldpair(k0 + h * 16,      k0 + h * 16 + 8);
    const bf16x16 b01 = ldpair(k0 + 32 + h * 16, k0 + 32 + h * 16 + 8);
    const bf16x16 b10 = ldpair(k1 + h * 16,      k1 + h * 16 + 8);
    const bf16x16 b11 = ldpair(k1 + 32 + h * 16, k1 + 32 + h * 16 + 8);
    f32x8 s0 = {}, s1 = {};
    s0 = wmma_bf16(aq0, b00, s0);
    s0 = wmma_bf16(aq1, b01, s0);
    s1 = wmma_bf16(aq0, b10, s1);
    s1 = wmma_bf16(aq1, b11, s1);

    // Online softmax. C layout: lane = column, row = r + 8*(lane/16); row
    // reductions are DPP ROW_XMASK trees within each 16-lane half (VALU only).
    float alpha[8];
#pragma unroll
    for (int r = 0; r < 8; ++r) {
      const float v0 = s0[r] * scale, v1 = s1[r] * scale;
      const float mx = rowmax16(fmaxf(v0, v1));
      const float mnew = fmaxf(mst[r], mx);
      const float a  = __expf(mst[r] - mnew);
      const float p0 = __expf(v0 - mnew);
      const float p1 = __expf(v1 - mnew);
      lst[r] = lst[r] * a + rowsum16(p0 + p1);
      mst[r] = mnew;
      alpha[r] = a;
      const int row = r + 8 * h;
      pst[row * 40 + l16]      = (__bf16)p0;   // C layout -> LDS row-major
      pst[row * 40 + 16 + l16] = (__bf16)p1;
    }
    asm volatile("s_wait_dscnt 0" ::: "memory");  // wave-private staging ready

#pragma unroll
    for (int t = 0; t < 4; ++t)
#pragma unroll
      for (int r = 0; r < 8; ++r) oacc[t][r] *= alpha[r];

    // P back in A-fragment layout (16 x 32, K = key-in-block).
    const __bf16* prow = pst + l16 * 40;
    const bf16x16 ap = ldpair(prow + h * 16, prow + h * 16 + 8);

    // O += P @ V from the shared V^T tile.
#pragma unroll
    for (int t = 0; t < 4; ++t) {
      const __bf16* vrp = &vt[buf][(t * 16 + l16) * 40];
      const bf16x16 bv = ldpair(vrp + h * 16, vrp + h * 16 + 8);
      oacc[t] = wmma_bf16(ap, bv, oacc[t]);
    }

    wait_async0();       // my async writes into the other buffer are done
    __syncthreads();     // everyone done reading cur + everyone's writes landed
  }

  // Normalize and write merged-head bf16 output.
#pragma unroll
  for (int r = 0; r < 8; ++r) {
    const float inv = 1.0f / lst[r];
    const int srow = qt * 16 + r + 8 * h;
    const size_t base =
        ((size_t)(bh / NH) * SEQ + srow) * DM + (size_t)(bh % NH) * DKH;
#pragma unroll
    for (int t = 0; t < 4; ++t)
      X2[base + t * 16 + l16] = (__bf16)(oacc[t][r] * inv);
  }
}

extern "C" void kernel_launch(void* const* d_in, const int* in_sizes, int n_in,
                              void* d_out, int out_size, void* d_ws, size_t ws_size,
                              hipStream_t stream) {
  (void)n_in; (void)out_size; (void)ws_size;
  const float* query = (const float*)d_in[0];
  const float* key   = (const float*)d_in[1];
  const float* value = (const float*)d_in[2];
  const float* w_q   = (const float*)d_in[3];
  const float* b_q   = (const float*)d_in[4];
  const float* w_k   = (const float*)d_in[5];
  const float* b_k   = (const float*)d_in[6];
  const float* w_v   = (const float*)d_in[7];
  const float* b_v   = (const float*)d_in[8];
  const float* w_o   = (const float*)d_in[9];
  const float* b_o   = (const float*)d_in[10];
  float* out = (float*)d_out;

  const int M  = in_sizes[0] / DM;  // B*S = 4096
  const int Bb = M / SEQ;           // 2

  // Workspace layout (bf16 elements; ~64 MB total).
  const size_t nx = (size_t)M * DM;       // activations
  const size_t nw = (size_t)DM * DM;      // weights
  __bf16* p   = (__bf16*)d_ws;
  __bf16* xq  = p;  p += nx;
  __bf16* xk  = p;  p += nx;
  __bf16* xv  = p;  p += nx;
  __bf16* wqb = p;  p += nw;
  __bf16* wkb = p;  p += nw;
  __bf16* wvb = p;  p += nw;
  __bf16* wob = p;  p += nw;
  __bf16* qb  = p;  p += nx;
  __bf16* kbf = p;  p += nx;
  __bf16* vbf = p;  p += nx;
  __bf16* x2  = p;  p += nx;

  dim3 blk(256);
  const unsigned gcx = (unsigned)((nx / 4 + 255) / 256);
  const unsigned gcw = (unsigned)((nw / 4 + 255) / 256);
  cvt_kernel<<<gcx, blk, 0, stream>>>(query, xq, (int)(nx / 4));
  cvt_kernel<<<gcx, blk, 0, stream>>>(key,   xk, (int)(nx / 4));
  cvt_kernel<<<gcx, blk, 0, stream>>>(value, xv, (int)(nx / 4));
  cvt_kernel<<<gcw, blk, 0, stream>>>(w_q, wqb, (int)(nw / 4));
  cvt_kernel<<<gcw, blk, 0, stream>>>(w_k, wkb, (int)(nw / 4));
  cvt_kernel<<<gcw, blk, 0, stream>>>(w_v, wvb, (int)(nw / 4));
  cvt_kernel<<<gcw, blk, 0, stream>>>(w_o, wob, (int)(nw / 4));

  // 32x64 tile per wave, 8 waves per block.
  dim3 gg((unsigned)((M / 32) * (DM / 64) / 8));
  gemm_kernel<0><<<gg, blk, 0, stream>>>(xq, wqb, b_q, qb,  nullptr);
  gemm_kernel<0><<<gg, blk, 0, stream>>>(xk, wkb, b_k, kbf, nullptr);
  gemm_kernel<1><<<gg, blk, 0, stream>>>(xv, wvb, b_v, vbf, nullptr);

  dim3 ga((unsigned)(Bb * NH * (SEQ / 128)));    // 128 q rows per block
  attn_kernel<<<ga, blk, 0, stream>>>(qb, kbf, vbf, x2);

  gemm_kernel<2><<<gg, blk, 0, stream>>>(x2, wob, b_o, nullptr, out);
}
